// QuantizedLinear158_69106023793067
// MI455X (gfx1250) — 8-bit matrix operand order repaired and checked by reading, compiled, not run on hardware
//
#include <hip/hip_runtime.h>

#define DIN  2048
#define DOUT 8192
#define MTOK 8192

typedef __attribute__((ext_vector_type(8))) int v8i;

union V8U {
    v8i v;
    int4 q[2];
    unsigned long long u[4];
};

#define WMMA_IU8(A, B, C) \
    __builtin_amdgcn_wmma_i32_16x16x64_iu8(true, (A), true, (B), (C), false, false)

__device__ __forceinline__ void load_a_frag(V8U& f, const signed char* p) {
    f.u[0] = *(const unsigned long long*)(p +  0);
    f.u[1] = *(const unsigned long long*)(p + 16);
    f.u[2] = *(const unsigned long long*)(p + 32);
    f.u[3] = *(const unsigned long long*)(p + 48);
}
__device__ __forceinline__ void load_b_frag(V8U& f, const signed char* p) {
    f.u[0] = *(const unsigned long long*)(p +  0);
    f.u[1] = *(const unsigned long long*)(p + 16);
    f.u[2] = *(const unsigned long long*)(p + 32);
    f.u[3] = *(const unsigned long long*)(p + 48);
}

__global__ void __launch_bounds__(256) bitnet_convert_w(const int* __restrict__ w,
                                                        signed char* __restrict__ w8)
{
    const int idx = blockIdx.x * 256 + threadIdx.x;
    const int4* w4 = (const int4*)w + (size_t)idx * 4;
    int packed[4];
#pragma unroll
    for (int c = 0; c < 4; ++c) {
        int4 t = w4[c];
        packed[c] = (t.x & 0xFF) | ((t.y & 0xFF) << 8) |
                    ((t.z & 0xFF) << 16) | (t.w << 24);
    }
    ((int4*)w8)[idx] = make_int4(packed[0], packed[1], packed[2], packed[3]);
}

__global__ void __launch_bounds__(256) bitnet_quant_rows(
        const float* __restrict__ x, const float* __restrict__ wscale,
        signed char* __restrict__ xq, float* __restrict__ rowScale)
{
    const int m   = blockIdx.x;
    const int tid = threadIdx.x;
    const float4* row = (const float4*)(x + (size_t)m * DIN);
    float4 a = row[tid * 2 + 0];
    float4 b = row[tid * 2 + 1];
    float mx = fmaxf(fmaxf(fmaxf(fabsf(a.x), fabsf(a.y)), fmaxf(fabsf(a.z), fabsf(a.w))),
                     fmaxf(fmaxf(fabsf(b.x), fabsf(b.y)), fmaxf(fabsf(b.z), fabsf(b.w))));

    __shared__ float red[256];
    red[tid] = mx;
    __syncthreads();
#pragma unroll
    for (int s = 128; s > 0; s >>= 1) {
        if (tid < s) red[tid] = fmaxf(red[tid], red[tid + s]);
        __syncthreads();
    }
    const float rowmax = fmaxf(red[0], 1e-5f);
    const float as = 127.0f / rowmax;

    const float va[8] = {a.x, a.y, a.z, a.w, b.x, b.y, b.z, b.w};
    int p0 = 0, p1 = 0;
#pragma unroll
    for (int i = 0; i < 4; ++i) {
        int q = (int)fminf(fmaxf(rintf(va[i] * as), -128.0f), 127.0f);
        p0 |= (q & 0xFF) << (8 * i);
    }
#pragma unroll
    for (int i = 0; i < 4; ++i) {
        int q = (int)fminf(fmaxf(rintf(va[4 + i] * as), -128.0f), 127.0f);
        p1 |= (q & 0xFF) << (8 * i);
    }
    ((int2*)(xq + (size_t)m * DIN))[tid] = make_int2(p0, p1);
    if (tid == 0) rowScale[m] = wscale[0] * rowmax * (1.0f / 127.0f);
}

__global__ void __launch_bounds__(256) bitnet_wmma_gemm(
        const signed char* __restrict__ Aq,
        const signed char* __restrict__ W8,
        const float* __restrict__ rowScale,
        const float* __restrict__ bias,
        float* __restrict__ out)
{
    const int NT   = DOUT / 64;
    const int wave = blockIdx.x * 8 + (threadIdx.x >> 5);
    const int lane = threadIdx.x & 31;
    const int nTile = wave & (NT - 1);
    const int mTile = wave >> 7;
    const int m0 = mTile * 32;
    const int n0 = nTile * 64;
    const int half = lane >> 4;
    const int l16  = lane & 15;

    const signed char* aP0 = Aq + (size_t)(m0 + l16) * DIN + half * 8;
    const signed char* aP1 = aP0 + (size_t)16 * DIN;
    const signed char* bP0 = W8 + (size_t)(n0 + l16) * DIN + half * 8;
    const signed char* bP1 = bP0 + (size_t)16 * DIN;
    const signed char* bP2 = bP0 + (size_t)32 * DIN;
    const signed char* bP3 = bP0 + (size_t)48 * DIN;

    v8i acc[2][4];
#pragma unroll
    for (int mi = 0; mi < 2; ++mi)
#pragma unroll
        for (int j = 0; j < 4; ++j) acc[mi][j] = (v8i){};

    V8U a0, a1, b0, b1, b2, b3;
    load_a_frag(a0, aP0);  load_a_frag(a1, aP1);
    load_b_frag(b0, bP0);  load_b_frag(b1, bP1);
    load_b_frag(b2, bP2);  load_b_frag(b3, bP3);

    for (int kk = 64; kk < DIN; kk += 64) {
        V8U na0, na1, nb0, nb1, nb2, nb3;
        load_a_frag(na0, aP0 + kk);  load_a_frag(na1, aP1 + kk);
        load_b_frag(nb0, bP0 + kk);  load_b_frag(nb1, bP1 + kk);
        load_b_frag(nb2, bP2 + kk);  load_b_frag(nb3, bP3 + kk);

        acc[0][0] = WMMA_IU8(a0.v, b0.v, acc[0][0]);
        acc[0][1] = WMMA_IU8(a0.v, b1.v, acc[0][1]);
        acc[0][2] = WMMA_IU8(a0.v, b2.v, acc[0][2]);
        acc[0][3] = WMMA_IU8(a0.v, b3.v, acc[0][3]);
        acc[1][0] = WMMA_IU8(a1.v, b0.v, acc[1][0]);
        acc[1][1] = WMMA_IU8(a1.v, b1.v, acc[1][1]);
        acc[1][2] = WMMA_IU8(a1.v, b2.v, acc[1][2]);
        acc[1][3] = WMMA_IU8(a1.v, b3.v, acc[1][3]);

        a0 = na0;  a1 = na1;
        b0 = nb0;  b1 = nb1;  b2 = nb2;  b3 = nb3;
    }
    acc[0][0] = WMMA_IU8(a0.v, b0.v, acc[0][0]);
    acc[0][1] = WMMA_IU8(a0.v, b1.v, acc[0][1]);
    acc[0][2] = WMMA_IU8(a0.v, b2.v, acc[0][2]);
    acc[0][3] = WMMA_IU8(a0.v, b3.v, acc[0][3]);
    acc[1][0] = WMMA_IU8(a1.v, b0.v, acc[1][0]);
    acc[1][1] = WMMA_IU8(a1.v, b1.v, acc[1][1]);
    acc[1][2] = WMMA_IU8(a1.v, b2.v, acc[1][2]);
    acc[1][3] = WMMA_IU8(a1.v, b3.v, acc[1][3]);

    const float bv0 = bias[n0 +  0 + l16];
    const float bv1 = bias[n0 + 16 + l16];
    const float bv2 = bias[n0 + 32 + l16];
    const float bv3 = bias[n0 + 48 + l16];
#pragma unroll
    for (int mi = 0; mi < 2; ++mi) {
#pragma unroll
        for (int r = 0; r < 8; ++r) {
            const int mrow = m0 + mi * 16 + half * 8 + r;
            const float sA = rowScale[mrow];
            float* o = out + (size_t)mrow * DOUT + n0 + l16;
            __builtin_nontemporal_store((float)acc[mi][0][r] * sA + bv0, o +  0);
            __builtin_nontemporal_store((float)acc[mi][1][r] * sA + bv1, o + 16);
            __builtin_nontemporal_store((float)acc[mi][2][r] * sA + bv2, o + 32);
            __builtin_nontemporal_store((float)acc[mi][3][r] * sA + bv3, o + 48);
        }
    }
}

extern "C" void kernel_launch(void* const* d_in, const int* in_sizes, int n_in,
                              void* d_out, int out_size, void* d_ws, size_t ws_size,
                              hipStream_t stream) {
    const float* x      = (const float*)d_in[0];
    const int*   wq     = (const int*)  d_in[1];
    const float* wscale = (const float*)d_in[2];
    const float* bias   = (const float*)d_in[3];
    float* out = (float*)d_out;

    signed char* W8 = (signed char*)d_ws;
    signed char* Xq = W8 + (size_t)DOUT * DIN;
    float* rowScale = (float*)(Xq + (size_t)MTOK * DIN);

    bitnet_convert_w<<<(DOUT * DIN / 16) / 256, 256, 0, stream>>>(wq, W8);
    bitnet_quant_rows<<<MTOK, 256, 0, stream>>>(x, wscale, Xq, rowScale);
    bitnet_wmma_gemm<<<(MTOK / 32) * (DOUT / 64) / 8, 256, 0, stream>>>(
        Xq, W8, rowScale, bias, out);
}
